// GroupTalkingAttention_74217034875258
// MI455X (gfx1250) — matrix operand order repaired and checked by reading, not compiled, not run on hardware
//
#include <hip/hip_runtime.h>
#include <hip/hip_bf16.h>

typedef _Float16 f16;
typedef __attribute__((ext_vector_type(16))) _Float16 v16h;
typedef __attribute__((ext_vector_type(8)))  float    v8f;

#define BATCH  8
#define NTOK   16384
#define CDIM   256
#define NHEADS 8
#define WSZ    8
#define GWIN   256   // windows per batch (16x16)
#define STOK   64    // tokens per window
#define DHEAD  32

// ---------------- WMMA fragment helpers (ISA 7.12.2 layouts) ----------------

__device__ __forceinline__ v8f wmma_f16(v16h a, v16h b, v8f c) {
  // D = A(16x32 f16) * B(32x16 f16) + C(16x16 f32)
  return __builtin_amdgcn_wmma_f32_16x16x32_f16(
      /*neg_a=*/false, a, /*neg_b=*/false, b,
      /*c_mod=*/(short)0, c, /*reuse_a=*/false, /*reuse_b=*/false);
}

// A fragment: rows m0..m0+15, cols k0..k0+31 of row-major f16 src (ld elements).
// Lane l: m = m0 + (l&15); kh = l>>4. VGPR j (pair):
//   j<4 : K = 2j + 8*kh ; j>=4 : K = 16 + 2(j-4) + 8*kh
__device__ __forceinline__ v16h frag_a(const f16* src, int ld, int m0, int k0, int lane) {
  const int m  = m0 + (lane & 15);
  const int kh = lane >> 4;
  const f16* p = src + m * ld + k0 + 8 * kh;
  v16h a;
#pragma unroll
  for (int j = 0; j < 4; ++j) {
    a[2 * j]     = p[2 * j];
    a[2 * j + 1] = p[2 * j + 1];
    a[8 + 2 * j]     = p[16 + 2 * j];
    a[8 + 2 * j + 1] = p[16 + 2 * j + 1];
  }
  return a;
}

// B fragment where column n of B is a contiguous row of src: B[k][n] = src[n*ld + k0 + k].
// Lane l: n = n0 + (l&15); kh = l>>4; elem j<8 : K = j + 8kh ; j>=8 : K = 16 + (j-8) + 8kh.
__device__ __forceinline__ v16h frag_b_rows(const f16* src, int ld, int n0, int k0, int lane) {
  const int n  = n0 + (lane & 15);
  const int kh = lane >> 4;
  const f16* p = src + n * ld + k0 + 8 * kh;
  v16h b;
#pragma unroll
  for (int j = 0; j < 16; ++j) b[j] = p[j < 8 ? j : j + 8];
  return b;
}

// B fragment from row-major [K][N] source: B[k][n] = src[(k0+k)*ld + n]
__device__ __forceinline__ v16h frag_b_cols(const f16* src, int ld, int k0, int n0, int lane) {
  const int n  = n0 + (lane & 15);
  const int kh = lane >> 4;
  const f16* p = src + (k0 + 8 * kh) * ld + n;
  v16h b;
#pragma unroll
  for (int j = 0; j < 16; ++j) b[j] = p[(j < 8 ? j : j + 8) * ld];
  return b;
}

// A fragment from the f32 attention cube [64][64][8]: elem (m,k) = attn[((q0+m)*64 + k0+k)*8 + h]
__device__ __forceinline__ v16h frag_a_attn(const float* attn, int h, int q0, int k0, int lane) {
  const int q  = q0 + (lane & 15);
  const int kh = lane >> 4;
  const float* p = attn + (q * 64 + k0 + 8 * kh) * 8 + h;
  v16h a;
#pragma unroll
  for (int j = 0; j < 4; ++j) {
    a[2 * j]     = (f16)p[(2 * j) * 8];
    a[2 * j + 1] = (f16)p[(2 * j + 1) * 8];
    a[8 + 2 * j]     = (f16)p[(16 + 2 * j) * 8];
    a[8 + 2 * j + 1] = (f16)p[(16 + 2 * j + 1) * 8];
  }
  return a;
}

// D store: lane l, VGPR r -> (m0 + 8*(l>>4) + r, n0 + (l&15))
__device__ __forceinline__ void store_d_f16(f16* dst, int ld, int m0, int n0, v8f c, int lane) {
  const int n  = n0 + (lane & 15);
  const int mb = m0 + 8 * (lane >> 4);
#pragma unroll
  for (int r = 0; r < 8; ++r) dst[(mb + r) * ld + n] = (f16)c[r];
}

__device__ __forceinline__ void store_d_attn(float* attn, int h, int q0, int k0, v8f c,
                                             float s, int lane) {
  const int n  = k0 + (lane & 15);
  const int qb = q0 + 8 * (lane >> 4);
#pragma unroll
  for (int r = 0; r < 8; ++r) attn[((qb + r) * 64 + n) * 8 + h] = c[r] * s;
}

// ---------------- weight f32 -> f16 prep kernel ----------------
// ws layout (f16): [0,131072) W_qk | [131072,196608) W_v | [196608,262144) W_proj
__global__ __launch_bounds__(256) void cvt_weights_kernel(
    const float* __restrict__ wqk, const float* __restrict__ wv,
    const float* __restrict__ wproj, f16* __restrict__ dst) {
  int i = blockIdx.x * 256 + threadIdx.x;
  if (i >= 262144) return;
  float v;
  if (i < 131072)      v = wqk[i];
  else if (i < 196608) v = wv[i - 131072];
  else                 v = wproj[i - 196608];
  dst[i] = (f16)v;
}

// ---------------- fused windowed talking-heads attention ----------------
// one workgroup (8 wave32s) per window; grid = B * GWIN = 2048
__global__ __launch_bounds__(256) void win_attn_kernel(
    const float* __restrict__ x, const float* __restrict__ query,
    const f16* __restrict__ wqk,    // [512][256] f16 (rows 0..255 = Wq, 256..511 = Wk)
    const f16* __restrict__ wv,     // [256][256] f16
    const f16* __restrict__ wproj,  // [256][256] f16
    const float* __restrict__ bproj,
    const float* __restrict__ wpre, const float* __restrict__ wpost,
    float* __restrict__ out) {
  extern __shared__ char smem[];
  f16*   sX    = (f16*)(smem);            // 64x256 f16: qw tile, then xw tile, then out tile
  f16*   sQ    = (f16*)(smem + 32768);    // 64x256
  f16*   sK    = (f16*)(smem + 65536);    // 64x256
  f16*   sV    = (f16*)(smem + 98304);    // 64x256
  float* sAttn = (float*)(smem + 131072); // [64][64][8] f32 = 128KB
  float* sWpre = (float*)(smem + 262144); // 8x8
  float* sWpost = sWpre + 64;             // 8x8

  const int tid  = threadIdx.x;
  const int wave = tid >> 5;
  const int lane = tid & 31;
  const int blk  = blockIdx.x;
  const int b    = blk >> 8;   // / GWIN
  const int g    = blk & 255;
  const int gy   = g >> 4, gx = g & 15;

  if (tid < 64) sWpre[tid] = wpre[tid];
  else if (tid < 128) sWpost[tid - 64] = wpost[tid - 64];

  // ---- load query window tile -> sX (f16), thread t = channel, s = row ----
#pragma unroll 4
  for (int s = 0; s < STOK; ++s) {
    int n = (gy * WSZ + (s >> 3)) * 128 + gx * WSZ + (s & 7);
    sX[s * CDIM + tid] = (f16)query[((size_t)b * NTOK + n) * CDIM + tid];
  }
  __syncthreads();

  // ---- q = qw @ Wq^T, k = qw @ Wk^T : 32 column strips, 4 per wave ----
  for (int rr = 0; rr < 4; ++rr) {
    const int strip = wave * 4 + rr;          // 0..31
    const int is_k  = strip >> 4;             // 0 -> q, 1 -> k
    const int n0    = (strip & 15) * 16;
    const f16* wsrc = wqk + (is_k ? (256 * 256) : 0);
    v8f acc[4] = {};
#pragma unroll
    for (int kt = 0; kt < 8; ++kt) {
      v16h bf = frag_b_rows(wsrc, CDIM, n0, kt * 32, lane);
#pragma unroll
      for (int mt = 0; mt < 4; ++mt) {
        v16h af = frag_a(sX, CDIM, mt * 16, kt * 32, lane);
        acc[mt] = wmma_f16(af, bf, acc[mt]);
      }
    }
    f16* dst = is_k ? sK : sQ;
#pragma unroll
    for (int mt = 0; mt < 4; ++mt) store_d_f16(dst, CDIM, mt * 16, n0, acc[mt], lane);
  }
  __syncthreads();

  // ---- load x window tile -> sX (overwrites qw) ----
#pragma unroll 4
  for (int s = 0; s < STOK; ++s) {
    int n = (gy * WSZ + (s >> 3)) * 128 + gx * WSZ + (s & 7);
    sX[s * CDIM + tid] = (f16)x[((size_t)b * NTOK + n) * CDIM + tid];
  }
  __syncthreads();

  // ---- v = xw @ Wv^T : 16 strips, 2 per wave ----
  for (int rr = 0; rr < 2; ++rr) {
    const int n0 = (wave * 2 + rr) * 16;
    v8f acc[4] = {};
#pragma unroll
    for (int kt = 0; kt < 8; ++kt) {
      v16h bf = frag_b_rows(wv, CDIM, n0, kt * 32, lane);
#pragma unroll
      for (int mt = 0; mt < 4; ++mt) {
        v16h af = frag_a(sX, CDIM, mt * 16, kt * 32, lane);
        acc[mt] = wmma_f16(af, bf, acc[mt]);
      }
    }
#pragma unroll
    for (int mt = 0; mt < 4; ++mt) store_d_f16(sV, CDIM, mt * 16, n0, acc[mt], lane);
  }
  __syncthreads();

  // ---- logits: attn[q][k][h] = scale * q_h @ k_h^T ; 128 16x16 tiles ----
  const float scale = 0.17677669529663687f;  // 1/sqrt(32)
  for (int t = wave; t < 128; t += 8) {
    const int h = t >> 4, qt = (t >> 2) & 3, kt2 = t & 3;
    v16h af = frag_a(sQ, CDIM, qt * 16, h * DHEAD, lane);
    v16h bf = frag_b_rows(sK, CDIM, kt2 * 16, h * DHEAD, lane);
    v8f acc = {};
    acc = wmma_f16(af, bf, acc);
    store_d_attn(sAttn, h, qt * 16, kt2 * 16, acc, scale, lane);
  }
  __syncthreads();

  // ---- pre-softmax talking heads: per (q,k), a <- Wpre @ a ----
#pragma unroll 2
  for (int i = 0; i < 16; ++i) {
    float* a = sAttn + (tid + 256 * i) * 8;
    float in[8], o[8];
#pragma unroll
    for (int h = 0; h < 8; ++h) in[h] = a[h];
#pragma unroll
    for (int h = 0; h < 8; ++h) {
      float s = 0.f;
#pragma unroll
      for (int j = 0; j < 8; ++j) s = fmaf(sWpre[h * 8 + j], in[j], s);
      o[h] = s;
    }
#pragma unroll
    for (int h = 0; h < 8; ++h) a[h] = o[h];
  }
  __syncthreads();

  // ---- softmax over key dim per (q,h): 512 tasks ----
  for (int i = 0; i < 2; ++i) {
    const int task = tid + 256 * i;
    const int q = task >> 3, h = task & 7;
    float* base = sAttn + q * 512 + h;
    float m = -1e30f;
#pragma unroll 8
    for (int k = 0; k < 64; ++k) m = fmaxf(m, base[k * 8]);
    float s = 0.f;
#pragma unroll 8
    for (int k = 0; k < 64; ++k) {
      float e = __expf(base[k * 8] - m);
      base[k * 8] = e;
      s += e;
    }
    const float inv = 1.f / s;
#pragma unroll 8
    for (int k = 0; k < 64; ++k) base[k * 8] *= inv;
  }
  __syncthreads();

  // ---- post-softmax talking heads ----
#pragma unroll 2
  for (int i = 0; i < 16; ++i) {
    float* a = sAttn + (tid + 256 * i) * 8;
    float in[8], o[8];
#pragma unroll
    for (int h = 0; h < 8; ++h) in[h] = a[h];
#pragma unroll
    for (int h = 0; h < 8; ++h) {
      float s = 0.f;
#pragma unroll
      for (int j = 0; j < 8; ++j) s = fmaf(sWpost[h * 8 + j], in[j], s);
      o[h] = s;
    }
#pragma unroll
    for (int h = 0; h < 8; ++h) a[h] = o[h];
  }
  __syncthreads();

  // ---- out_h = attn_h @ v_h -> sX reused as [64][256] f16 output tile ----
  for (int t = wave; t < 64; t += 8) {
    const int h = t >> 3, qt = (t >> 1) & 3, nt = t & 1;
    v8f acc = {};
#pragma unroll
    for (int ks = 0; ks < 2; ++ks) {
      v16h af = frag_a_attn(sAttn, h, qt * 16, ks * 32, lane);
      v16h bf = frag_b_cols(sV, CDIM, ks * 32, h * DHEAD + nt * 16, lane);
      acc = wmma_f16(af, bf, acc);
    }
    store_d_f16(sX, CDIM, qt * 16, h * DHEAD + nt * 16, acc, lane);
  }
  __syncthreads();

  // ---- final projection: out = sX @ Wproj^T + b, scatter to global (un-window) ----
  for (int rr = 0; rr < 2; ++rr) {
    const int n0 = (wave * 2 + rr) * 16;
    v8f acc[4] = {};
#pragma unroll
    for (int kt = 0; kt < 8; ++kt) {
      v16h bf = frag_b_rows(wproj, CDIM, n0, kt * 32, lane);
#pragma unroll
      for (int mt = 0; mt < 4; ++mt) {
        v16h af = frag_a(sX, CDIM, mt * 16, kt * 32, lane);
        acc[mt] = wmma_f16(af, bf, acc[mt]);
      }
    }
    const int o = n0 + (lane & 15);
    const float bias = bproj[o];
#pragma unroll
    for (int mt = 0; mt < 4; ++mt) {
      const int sbase = mt * 16 + 8 * (lane >> 4);
#pragma unroll
      for (int r = 0; r < 8; ++r) {
        const int s = sbase + r;
        const int n = (gy * WSZ + (s >> 3)) * 128 + gx * WSZ + (s & 7);
        out[((size_t)b * NTOK + n) * CDIM + o] = acc[mt][r] + bias;
      }
    }
  }
}

// ---------------- host launch ----------------
extern "C" void kernel_launch(void* const* d_in, const int* in_sizes, int n_in,
                              void* d_out, int out_size, void* d_ws, size_t ws_size,
                              hipStream_t stream) {
  (void)in_sizes; (void)n_in; (void)out_size; (void)ws_size;
  const float* x      = (const float*)d_in[0];
  const float* query  = (const float*)d_in[1];
  const float* W_qk   = (const float*)d_in[2];
  const float* W_v    = (const float*)d_in[3];
  const float* W_proj = (const float*)d_in[4];
  const float* b_proj = (const float*)d_in[5];
  const float* W_pre  = (const float*)d_in[6];
  const float* W_post = (const float*)d_in[7];
  float* out = (float*)d_out;

  f16* wf16 = (f16*)d_ws;  // 262144 f16 = 512 KB scratch
  cvt_weights_kernel<<<1024, 256, 0, stream>>>(W_qk, W_v, W_proj, wf16);

  const int smem_bytes = 262656;  // 256.5 KB dynamic LDS (<= 320 KB WGP limit)
  hipFuncSetAttribute((const void*)win_attn_kernel,
                      hipFuncAttributeMaxDynamicSharedMemorySize, smem_bytes);
  win_attn_kernel<<<BATCH * GWIN, 256, smem_bytes, stream>>>(
      x, query,
      /*wqk=*/wf16,
      /*wv=*/wf16 + 131072,
      /*wproj=*/wf16 + 196608,
      b_proj, W_pre, W_post, out);
}
